// GlideHybridAttention_2568390443436
// MI455X (gfx1250) — matrix operand order repaired and checked by reading, not compiled, not run on hardware
//
#include <hip/hip_runtime.h>

// ---------------------------------------------------------------------------
// GlideHybridAttention for MI455X (gfx1250, wave32, WMMA)
// GEMMs + sliding-window attention use v_wmma_f32_16x16x32_bf16.
// GEMM A-tile staging uses async global->LDS (ASYNCcnt) with double-buffered
// LDS pipelining; B tiles are vector-loaded and transpose-staged post-compute.
// GLA scan is a column-parallel f32 recurrence (not a matmul shape).
// ---------------------------------------------------------------------------

typedef __attribute__((ext_vector_type(16))) __bf16 v16bf;
typedef __attribute__((ext_vector_type(8)))  float  v8f;
typedef __attribute__((ext_vector_type(4)))  __bf16 v4bf;

#define CN    2048      // sequence length
#define CHID  4096      // hidden size
#define CH    32        // query heads
#define CKVH  8         // kv heads
#define CD    128       // head dim
#define CW    64        // window

__device__ __forceinline__ v8f v8f_zero() {
  v8f z;
#pragma unroll
  for (int i = 0; i < 8; ++i) z[i] = 0.0f;
  return z;
}

// async copy of 16 bytes global -> LDS (per lane), tracked by ASYNCcnt
__device__ __forceinline__ void async_copy_b128(const void* gptr, void* lptr) {
  unsigned lds = (unsigned)(uintptr_t)lptr;   // low 32 bits of generic = LDS offset
  asm volatile("global_load_async_to_lds_b128 %0, %1, off"
               :: "v"(lds), "v"(gptr) : "memory");
}
__device__ __forceinline__ void wait_async0() {
#if __has_builtin(__builtin_amdgcn_s_wait_asynccnt)
  __builtin_amdgcn_s_wait_asynccnt(0);
#else
  asm volatile("s_wait_asynccnt 0" ::: "memory");
#endif
}

// A-matrix fragment (16x32 bf16), tile stored row-major [m][k] with given k-stride.
// lanes 0-15 -> row=lane, K {0..7, 16..23}; lanes 16-31 -> K {8..15, 24..31}.
__device__ __forceinline__ v16bf frag_a(const __bf16* base, int stride) {
  int l = threadIdx.x & 31;
  const __bf16* p = base + (l & 15) * stride + ((l >> 4) << 3);
  v16bf f;
#pragma unroll
  for (int j = 0; j < 8; ++j) { f[j] = p[j]; f[j + 8] = p[j + 16]; }
  return f;
}

// B-matrix fragment (32x16 bf16), tile stored as [n][k] (B transposed) with k-stride.
// lanes 0-15 hold K {0..7, 16..23}, lanes 16-31 hold K {8..15, 24..31}, col = lane%16.
__device__ __forceinline__ v16bf frag_b(const __bf16* base, int stride) {
  int l = threadIdx.x & 31;
  const __bf16* p = base + (l & 15) * stride + ((l >> 4) << 3);
  v16bf f;
#pragma unroll
  for (int j = 0; j < 8; ++j) { f[j] = p[j]; f[j + 8] = p[j + 16]; }
  return f;
}

#define WMMA_BF16(a, b, c) \
  __builtin_amdgcn_wmma_f32_16x16x32_bf16(false, (a), false, (b), (short)0, (c), false, false)

// ---------------------------------------------------------------------------
// f32 -> bf16 conversion (vectorized: 4 elems/thread)
// ---------------------------------------------------------------------------
__global__ __launch_bounds__(256) void cvt_bf16_kernel(const float* __restrict__ in,
                                                       __bf16* __restrict__ out, int n) {
  int i4 = (blockIdx.x * 256 + threadIdx.x) * 4;
  if (i4 < n) {
    float4 t = *(const float4*)&in[i4];
    v4bf o;
    o[0] = (__bf16)t.x; o[1] = (__bf16)t.y; o[2] = (__bf16)t.z; o[3] = (__bf16)t.w;
    *(v4bf*)&out[i4] = o;
  }
}

// ---------------------------------------------------------------------------
// C[f32, MxN] = A[bf16, MxK, row-major] @ B[bf16, KxN, row-major]
// workgroup tile 128x128, 8 waves (32x64 each), K-step 64, double-buffered LDS.
// Pipeline per iter: wait(cur DMA) -> barrier -> issue async A(next) + global
// B(next) loads -> WMMA on cur -> transpose-store B(next) to LDS.
// ---------------------------------------------------------------------------
__global__ __launch_bounds__(256) void gemm_bf16_kernel(const __bf16* __restrict__ A,
                                                        const __bf16* __restrict__ B,
                                                        float* __restrict__ C,
                                                        int M, int N, int K) {
  constexpr int BK = 64;
  constexpr int LDK = BK + 8;   // 72 elems = 144B rows: keeps 16B alignment
  __shared__ __align__(16) __bf16 As[2][128 * LDK];   // [m][k]
  __shared__ __align__(16) __bf16 Bs[2][128 * LDK];   // [n][k]  (B transposed tile)

  const int tile_m = blockIdx.y * 128;
  const int tile_n = blockIdx.x * 128;
  const int tid = threadIdx.x;
  const int lane = tid & 31;
  const int w = tid >> 5;
  const int wr = (w & 3) * 32;    // wave row offset
  const int wc = (w >> 2) * 64;   // wave col offset

  v8f acc[2][4];
#pragma unroll
  for (int rt = 0; rt < 2; ++rt)
#pragma unroll
    for (int ct = 0; ct < 4; ++ct) acc[rt][ct] = v8f_zero();

  auto stage_a = [&](int buf, int k0) {
#pragma unroll
    for (int i = 0; i < 4; ++i) {
      int idx8 = tid + i * 256;                 // 1024 16B chunks of 128x64 tile
      int r = idx8 >> 3, c8 = (idx8 & 7) * 8;
      async_copy_b128(&A[(size_t)(tile_m + r) * K + (k0 + c8)],
                      &As[buf][r * LDK + c8]);
    }
  };
  auto load_b = [&](int k0, uint4* t) {
#pragma unroll
    for (int i = 0; i < 4; ++i) {
      int idx8 = tid + i * 256;
      int kk = idx8 >> 4, n8 = (idx8 & 15) * 8; // coalesced 16B read along n
      t[i] = *(const uint4*)&B[(size_t)(k0 + kk) * N + (tile_n + n8)];
    }
  };
  auto store_b = [&](int buf, const uint4* t) {
#pragma unroll
    for (int i = 0; i < 4; ++i) {
      int idx8 = tid + i * 256;
      int kk = idx8 >> 4, n8 = (idx8 & 15) * 8;
      const __bf16* tb = (const __bf16*)&t[i];
#pragma unroll
      for (int j = 0; j < 8; ++j) Bs[buf][(n8 + j) * LDK + kk] = tb[j];
    }
  };

  uint4 breg[4];
  // prologue: fully stage tile 0
  stage_a(0, 0);
  load_b(0, breg);
  store_b(0, breg);

  const int niter = K / BK;
  for (int it = 0; it < niter; ++it) {
    const int cur = it & 1, nxt = cur ^ 1;
    const int k0 = it * BK;
    const bool more = (it + 1) < niter;

    wait_async0();        // cur tile's DMA (this wave) complete
    __syncthreads();      // all waves: cur tile resident, prev reads retired

    if (more) {
      stage_a(nxt, k0 + BK);        // async DMA overlaps compute below
      load_b(k0 + BK, breg);        // global loads in flight during compute
      __builtin_prefetch(&A[(size_t)(tile_m + (tid >> 1)) * K + (k0 + 2 * BK)], 0, 0);
    }

#pragma unroll
    for (int kk = 0; kk < BK; kk += 32) {
      v16bf a0 = frag_a(&As[cur][wr * LDK + kk], LDK);
      v16bf a1 = frag_a(&As[cur][(wr + 16) * LDK + kk], LDK);
#pragma unroll
      for (int ct = 0; ct < 4; ++ct) {
        v16bf b = frag_b(&Bs[cur][(wc + ct * 16) * LDK + kk], LDK);
        acc[0][ct] = WMMA_BF16(a0, b, acc[0][ct]);
        acc[1][ct] = WMMA_BF16(a1, b, acc[1][ct]);
      }
    }

    if (more) store_b(nxt, breg);   // transpose-stage B after compute
  }

  const int cl = lane & 15;
  const int rh = (lane >> 4) * 8;
#pragma unroll
  for (int rt = 0; rt < 2; ++rt)
#pragma unroll
    for (int ct = 0; ct < 4; ++ct)
#pragma unroll
      for (int e = 0; e < 8; ++e) {
        int r = tile_m + wr + rt * 16 + rh + e;
        int c = tile_n + wc + ct * 16 + cl;
        C[(size_t)r * N + c] = acc[rt][ct][e];
      }
}

// ---------------------------------------------------------------------------
// per (n, head) query prep: softmax over D -> qs ; RoPE(pre-softmax) -> rq (bf16)
// ---------------------------------------------------------------------------
__global__ __launch_bounds__(128) void prep_q_kernel(const float* __restrict__ qf,
                                                     float* __restrict__ qs,
                                                     __bf16* __restrict__ rq) {
  const int n = blockIdx.x, h = blockIdx.y, d = threadIdx.x;
  const size_t base = ((size_t)n * CH + h) * CD;
  float x = qf[base + d];
  __shared__ float red[128];

  red[d] = x;
  __syncthreads();
#pragma unroll
  for (int s = 64; s > 0; s >>= 1) {
    if (d < s) red[d] = fmaxf(red[d], red[d + s]);
    __syncthreads();
  }
  float mx = red[0];
  __syncthreads();
  float e = expf(x - mx);
  red[d] = e;
  __syncthreads();
#pragma unroll
  for (int s = 64; s > 0; s >>= 1) {
    if (d < s) red[d] += red[d + s];
    __syncthreads();
  }
  qs[base + d] = e / red[0];

  int dh = d & 63;
  float inv = expf(-(float)dh * (logf(10000.0f) / 64.0f));
  float fr = (float)n * inv;
  float cs = cosf(fr), sn = sinf(fr);
  float other = qf[base + ((d < 64) ? d + 64 : d - 64)];
  float rot = (d < 64) ? -other : other;
  rq[base + d] = (__bf16)(x * cs + rot * sn);
}

// ---------------------------------------------------------------------------
// per (n, kv head): softmax(k) -> ks ; exp(logsigmoid(k)/16) -> eg ;
// RoPE(k) -> rk (bf16) ; v -> vb (bf16)
// ---------------------------------------------------------------------------
__global__ __launch_bounds__(128) void prep_k_kernel(const float* __restrict__ kf,
                                                     const float* __restrict__ vf,
                                                     float* __restrict__ ks,
                                                     float* __restrict__ eg,
                                                     __bf16* __restrict__ rk,
                                                     __bf16* __restrict__ vb) {
  const int n = blockIdx.x, h = blockIdx.y, d = threadIdx.x;
  const size_t base = ((size_t)n * CKVH + h) * CD;
  float x = kf[base + d];
  __shared__ float red[128];

  red[d] = x;
  __syncthreads();
#pragma unroll
  for (int s = 64; s > 0; s >>= 1) {
    if (d < s) red[d] = fmaxf(red[d], red[d + s]);
    __syncthreads();
  }
  float mx = red[0];
  __syncthreads();
  float e = expf(x - mx);
  red[d] = e;
  __syncthreads();
#pragma unroll
  for (int s = 64; s > 0; s >>= 1) {
    if (d < s) red[d] += red[d + s];
    __syncthreads();
  }
  ks[base + d] = e / red[0];

  float ls = fminf(x, 0.0f) - log1pf(expf(-fabsf(x)));
  eg[base + d] = expf(ls * (1.0f / 16.0f));

  int dh = d & 63;
  float inv = expf(-(float)dh * (logf(10000.0f) / 64.0f));
  float fr = (float)n * inv;
  float cs = cosf(fr), sn = sinf(fr);
  float other = kf[base + ((d < 64) ? d + 64 : d - 64)];
  float rot = (d < 64) ? -other : other;
  rk[base + d] = (__bf16)(x * cs + rot * sn);

  vb[base + d] = (__bf16)vf[base + d];
}

// ---------------------------------------------------------------------------
// GLA scan: grid (H, 8 v-groups of 16 cols), 256 threads. Thread owns 1 v-column
// x 8 k-rows of S. S[k,v] <- S*eg[k] + ks[k]*v[v] ; o[v] = sum_k q[k]*S[k,v]
// reduced over 16 row-segments. 8-step chunks staged in LDS.
// ---------------------------------------------------------------------------
__global__ __launch_bounds__(256) void gla_scan_kernel(const float* __restrict__ qs,
                                                       const float* __restrict__ ks,
                                                       const float* __restrict__ eg,
                                                       const float* __restrict__ vf,
                                                       float* __restrict__ olin) {
  const int h = blockIdx.x, vg = blockIdx.y;
  const int kvh = h >> 2;
  const int tid = threadIdx.x;
  const int col = tid & 15;       // local v column (16 per workgroup)
  const int seg = tid >> 4;       // 0..15 -> k rows seg*8..seg*8+7

  __shared__ float qb[8][128], kb[8][128], gb[8][128], vbuf[8][16];
  __shared__ float part[8][16][16];   // [t][seg][col]

  float S[8];
#pragma unroll
  for (int i = 0; i < 8; ++i) S[i] = 0.0f;

  for (int t0 = 0; t0 < CN; t0 += 8) {
    {
      // each of qb/kb/gb is 1024 floats = 256 threads x one float4
      int tt = tid >> 5, dd = (tid & 31) * 4;
      size_t qo = ((size_t)(t0 + tt) * CH + h) * CD + dd;
      size_t ko = ((size_t)(t0 + tt) * CKVH + kvh) * CD + dd;
      *(float4*)&qb[tt][dd] = *(const float4*)&qs[qo];
      *(float4*)&kb[tt][dd] = *(const float4*)&ks[ko];
      *(float4*)&gb[tt][dd] = *(const float4*)&eg[ko];
      if (tid < 128) {
        int t2 = tid >> 4, c2 = tid & 15;
        vbuf[t2][c2] = vf[((size_t)(t0 + t2) * CKVH + kvh) * CD + vg * 16 + c2];
      }
    }
    __syncthreads();

#pragma unroll
    for (int t = 0; t < 8; ++t) {
      float vv = vbuf[t][col];
      float acc = 0.0f;
#pragma unroll
      for (int i = 0; i < 8; ++i) {
        int r = seg * 8 + i;
        float s = S[i] * gb[t][r] + kb[t][r] * vv;
        S[i] = s;
        acc += qb[t][r] * s;
      }
      part[t][seg][col] = acc;
    }
    __syncthreads();

    if (tid < 128) {
      int tt = tid >> 4, cc = tid & 15;
      float o = 0.0f;
#pragma unroll
      for (int s2 = 0; s2 < 16; ++s2) o += part[tt][s2][cc];
      olin[((size_t)h * CN + (t0 + tt)) * CD + vg * 16 + cc] = o;
    }
    __syncthreads();
  }
}

// ---------------------------------------------------------------------------
// Sliding-window attention (blocked: [prev block, own block]) + 0.5/0.5 combine.
// grid (H, NB=32). WMMA for QK^T and PV. LDS = 64KB, buffers reused.
// ---------------------------------------------------------------------------
__global__ __launch_bounds__(256) void swa_kernel(const __bf16* __restrict__ rq,
                                                  const __bf16* __restrict__ rk,
                                                  const __bf16* __restrict__ vb,
                                                  const float* __restrict__ olin,
                                                  __bf16* __restrict__ ocomb) {
  __shared__ __align__(16) __bf16 Qs[64 * 128];   // Q (phase 1) then P (phase 2)
  __shared__ __align__(16) __bf16 Kb[64 * 128];   // K half (phase 1) then V^T half (phase 2)
  __shared__ float  Sc[64 * 128];                 // scores

  const int h = blockIdx.x, blk = blockIdx.y;
  const int kvh = h >> 2;
  const int tid = threadIdx.x;
  const int lane = tid & 31;
  const int w = tid >> 5;
  const int cl = lane & 15;
  const int rh = (lane >> 4) * 8;

  // load Q block [64 x 128] : 16B chunks
#pragma unroll
  for (int i = 0; i < 4; ++i) {
    int idx8 = tid + i * 256;
    int r = idx8 >> 4, d8 = (idx8 & 15) * 8;
    int n = blk * 64 + r;
    *(uint4*)&Qs[r * 128 + d8] = *(const uint4*)&rq[((size_t)n * CH + h) * CD + d8];
  }

  // ---- scores = QK^T (keys streamed in 2 halves of 64) ----
  const int srow = (w & 3) * 16;
  const int scol = (w >> 2) * 32;
  for (int kh = 0; kh < 2; ++kh) {
    __syncthreads();
#pragma unroll
    for (int i = 0; i < 4; ++i) {
      int idx8 = tid + i * 256;
      int r = idx8 >> 4, d8 = (idx8 & 15) * 8;   // r = local key
      int n = (blk - 1) * 64 + kh * 64 + r;
      uint4 t = make_uint4(0u, 0u, 0u, 0u);
      if (n >= 0) t = *(const uint4*)&rk[((size_t)n * CKVH + kvh) * CD + d8];
      *(uint4*)&Kb[r * 128 + d8] = t;
    }
    __syncthreads();

    v8f sacc[2] = {v8f_zero(), v8f_zero()};
#pragma unroll
    for (int kk = 0; kk < 128; kk += 32) {
      v16bf a = frag_a(&Qs[srow * 128 + kk], 128);
#pragma unroll
      for (int ct = 0; ct < 2; ++ct) {
        v16bf b = frag_b(&Kb[(scol + ct * 16) * 128 + kk], 128);
        sacc[ct] = WMMA_BF16(a, b, sacc[ct]);
      }
    }
#pragma unroll
    for (int ct = 0; ct < 2; ++ct)
#pragma unroll
      for (int e = 0; e < 8; ++e) {
        int r = srow + rh + e;
        int c = kh * 64 + scol + ct * 16 + cl;   // key offset in concat of 2 blocks
        bool valid = (c >= r) && (c <= r + CW) && (blk > 0 || c >= 64);
        Sc[r * 128 + c] = valid ? sacc[ct][e] * 0.08838834764831845f : -3.0e30f;
      }
  }
  __syncthreads();

  // ---- softmax rows, P -> Qs buffer (bf16) ----
  if (tid < 64) {
    int r = tid;
    float mx = -3.0e30f;
    for (int c = 0; c < 128; ++c) mx = fmaxf(mx, Sc[r * 128 + c]);
    float sum = 0.0f;
    for (int c = 0; c < 128; ++c) {
      float e = expf(Sc[r * 128 + c] - mx);
      Sc[r * 128 + c] = e;
      sum += e;
    }
    float inv = 1.0f / sum;
    for (int c = 0; c < 128; ++c) Qs[r * 128 + c] = (__bf16)(Sc[r * 128 + c] * inv);
  }

  // ---- out = P @ V (V^T halves streamed into Kb) ----
  const int orow = (w & 3) * 16;
  const int ocol = (w >> 2) * 64;
  v8f oacc[4] = {v8f_zero(), v8f_zero(), v8f_zero(), v8f_zero()};
  for (int kh = 0; kh < 2; ++kh) {
    __syncthreads();
#pragma unroll
    for (int i = 0; i < 4; ++i) {
      int idx8 = tid + i * 256;
      int key = idx8 >> 4, d8 = (idx8 & 15) * 8;  // coalesced 16B read along d
      int n = (blk - 1) * 64 + kh * 64 + key;
      uint4 t = make_uint4(0u, 0u, 0u, 0u);
      if (n >= 0) t = *(const uint4*)&vb[((size_t)n * CKVH + kvh) * CD + d8];
      const __bf16* tb = (const __bf16*)&t;
#pragma unroll
      for (int j = 0; j < 8; ++j) Kb[(d8 + j) * 64 + key] = tb[j];
    }
    __syncthreads();
#pragma unroll
    for (int kk = 0; kk < 64; kk += 32) {
      v16bf a = frag_a(&Qs[orow * 128 + kh * 64 + kk], 128);
#pragma unroll
      for (int ct = 0; ct < 4; ++ct) {
        v16bf b = frag_b(&Kb[(ocol + ct * 16) * 64 + kk], 64);
        oacc[ct] = WMMA_BF16(a, b, oacc[ct]);
      }
    }
  }

  // ---- combine 0.5*y + 0.5*o_lin, cast bf16, store [n][h*D+d] ----
#pragma unroll
  for (int ct = 0; ct < 4; ++ct)
#pragma unroll
    for (int e = 0; e < 8; ++e) {
      int r = orow + rh + e;
      int n = blk * 64 + r;
      int d = ocol + ct * 16 + cl;
      float y = oacc[ct][e];
      float ol = olin[((size_t)h * CN + n) * CD + d];
      ocomb[(size_t)n * (CH * CD) + h * CD + d] = (__bf16)(0.5f * y + 0.5f * ol);
    }
}

// ---------------------------------------------------------------------------
// launch
// ---------------------------------------------------------------------------
extern "C" void kernel_launch(void* const* d_in, const int* in_sizes, int n_in,
                              void* d_out, int out_size, void* d_ws, size_t ws_size,
                              hipStream_t stream) {
  (void)in_sizes; (void)n_in; (void)out_size; (void)ws_size;
  const float* hs = (const float*)d_in[0];
  const float* Wq = (const float*)d_in[1];
  const float* Wk = (const float*)d_in[2];
  const float* Wv = (const float*)d_in[3];
  const float* Wo = (const float*)d_in[4];
  float* out = (float*)d_out;

  char* ws = (char*)d_ws;
  size_t off = 0;
  auto take = [&](size_t bytes) -> void* {
    void* p = ws + off;
    off += (bytes + 255) & ~(size_t)255;
    return p;
  };

  __bf16* hs_bf = (__bf16*)take((size_t)CN * CHID * 2);
  __bf16* Wq_bf = (__bf16*)take((size_t)CHID * CHID * 2);
  __bf16* Wk_bf = (__bf16*)take((size_t)CHID * (CKVH * CD) * 2);
  __bf16* Wv_bf = (__bf16*)take((size_t)CHID * (CKVH * CD) * 2);
  __bf16* Wo_bf = (__bf16*)take((size_t)CHID * CHID * 2);
  float*  q_f   = (float*)take((size_t)CN * CH * CD * 4);
  float*  k_f   = (float*)take((size_t)CN * CKVH * CD * 4);
  float*  v_f   = (float*)take((size_t)CN * CKVH * CD * 4);
  float*  qs_f  = (float*)take((size_t)CN * CH * CD * 4);
  float*  ks_f  = (float*)take((size_t)CN * CKVH * CD * 4);
  float*  eg_f  = (float*)take((size_t)CN * CKVH * CD * 4);
  __bf16* rq_bf = (__bf16*)take((size_t)CN * CH * CD * 2);
  __bf16* rk_bf = (__bf16*)take((size_t)CN * CKVH * CD * 2);
  __bf16* vb_bf = (__bf16*)take((size_t)CN * CKVH * CD * 2);
  float*  olin  = (float*)take((size_t)CH * CN * CD * 4);
  __bf16* ocomb = (__bf16*)take((size_t)CN * CH * CD * 2);

  auto cvt = [&](const float* src, __bf16* dst, int n) {
    cvt_bf16_kernel<<<(n / 4 + 255) / 256, 256, 0, stream>>>(src, dst, n);
  };
  cvt(hs, hs_bf, CN * CHID);
  cvt(Wq, Wq_bf, CHID * CHID);
  cvt(Wk, Wk_bf, CHID * CKVH * CD);
  cvt(Wv, Wv_bf, CHID * CKVH * CD);
  cvt(Wo, Wo_bf, CHID * CHID);

  // projections
  gemm_bf16_kernel<<<dim3((CH * CD) / 128, CN / 128), 256, 0, stream>>>(
      hs_bf, Wq_bf, q_f, CN, CH * CD, CHID);
  gemm_bf16_kernel<<<dim3((CKVH * CD) / 128, CN / 128), 256, 0, stream>>>(
      hs_bf, Wk_bf, k_f, CN, CKVH * CD, CHID);
  gemm_bf16_kernel<<<dim3((CKVH * CD) / 128, CN / 128), 256, 0, stream>>>(
      hs_bf, Wv_bf, v_f, CN, CKVH * CD, CHID);

  // per-head prep
  prep_q_kernel<<<dim3(CN, CH), 128, 0, stream>>>(q_f, qs_f, rq_bf);
  prep_k_kernel<<<dim3(CN, CKVH), 128, 0, stream>>>(k_f, v_f, ks_f, eg_f, rk_bf, vb_bf);

  // gated linear attention scan
  gla_scan_kernel<<<dim3(CH, 8), 256, 0, stream>>>(qs_f, ks_f, eg_f, v_f, olin);

  // sliding-window attention + combine
  swa_kernel<<<dim3(CH, CN / CW), 256, 0, stream>>>(rq_bf, rk_bf, vb_bf, olin, ocomb);

  // output projection
  gemm_bf16_kernel<<<dim3(CHID / 128, CN / 128), 256, 0, stream>>>(
      ocomb, Wo_bf, out, CN, CHID, CHID);
}
